// GTV_66159676227632
// MI455X (gfx1250) — matrix operand order repaired and checked by reading, not compiled, not run on hardware
//
#include <hip/hip_runtime.h>
#include <hip/hip_bf16.h>
#include <math.h>

#define WIDTH 36
#define NN    (WIDTH * WIDTH)          // 1296 nodes
#define PW    (WIDTH + 2)              // 38 padded width (halo)
#define PNN   (PW * PW)                // 1444 padded pixels
#define EH    (WIDTH * (WIDTH - 1))    // 1260 horizontal edges (first in idx order)
#define EV    ((WIDTH - 1) * WIDTH)    // 1260 vertical edges
#define NE    (EH + EV)                // 2520
#define BATCH 2
#define CH    3
#define SIGMA2_INV 10000.0f            // 1 / 0.01^2
#define ZMAXV 0.01f
#define NITERS 9
#define CG_MAX 500

typedef __bf16 bf16;
typedef __attribute__((ext_vector_type(16))) __bf16 v16bf;
typedef __attribute__((ext_vector_type(8)))  __bf16 v8bf;
typedef __attribute__((ext_vector_type(8)))  float  v8f;

// padded pixel-major activation index: [b][y+1][x+1][ic], channel contiguous
__device__ __forceinline__ int act_idx(int b, int py, int px) {
    return (b * PNN + (py + 1) * PW + (px + 1)) * 32;
}

// ---------------------------------------------------------------------------
// Zero both padded activation buffers (contiguous in ws) so halos read 0.
// ---------------------------------------------------------------------------
__global__ void zero_kernel(unsigned int* __restrict__ p, int nwords) {
    int id = blockIdx.x * blockDim.x + threadIdx.x;
    if (id < nwords) p[id] = 0u;
}

// ---------------------------------------------------------------------------
// Pack mid-layer weights f32 [l][oc][ic][3][3] into per-lane WMMA B fragments:
// dst[l][tap][nt][lane][e] bf16, so conv_mid loads one aligned 32B chunk.
// B layout (32x16 bf16): lane half hi -> K = 8*hi + e (e<8), 16 + 8*hi + e-8 (e>=8), N = lane&15.
// ---------------------------------------------------------------------------
__global__ void pack_mid_w(const float* __restrict__ w, bf16* __restrict__ dst) {
    int id = blockIdx.x * blockDim.x + threadIdx.x;
    if (id >= 5 * 9216) return;
    int l   = id / 9216;
    int r   = id % 9216;
    int t   = r / 1024;            // tap 0..8
    int r2  = r % 1024;
    int nt  = r2 / 512;            // n-tile 0..1
    int li  = r2 % 512;
    int lane = li / 16, e = li % 16;
    int mrow = lane & 15, hi = lane >> 4;
    int oc = nt * 16 + mrow;
    int ic = (e < 8) ? (8 * hi + e) : (16 + 8 * hi + e - 8);
    dst[id] = (bf16)w[l * 9216 + (oc * 32 + ic) * 9 + t];
}

// Pack output-layer weights [6][32][3][3] -> dst[tap][lane][e], oc>=6 zero.
__global__ void pack_out_w(const float* __restrict__ w, bf16* __restrict__ dst) {
    int id = blockIdx.x * blockDim.x + threadIdx.x;
    if (id >= 9 * 512) return;
    int t  = id / 512;
    int li = id % 512;
    int lane = li / 16, e = li % 16;
    int oc = lane & 15, hi = lane >> 4;
    int ic = (e < 8) ? (8 * hi + e) : (16 + 8 * hi + e - 8);
    dst[id] = (oc < 6) ? (bf16)w[(oc * 32 + ic) * 9 + t] : (bf16)0.0f;
}

// ---------------------------------------------------------------------------
// Layer 1: 3->32 direct conv (K=27 too thin for WMMA), ReLU, bf16 padded out.
// ---------------------------------------------------------------------------
__global__ void conv_in_kernel(const float* __restrict__ xf,
                               const float* __restrict__ w,     // [32,3,3,3]
                               const float* __restrict__ bias,  // [32]
                               bf16* __restrict__ actOut) {     // padded
    int id = blockIdx.x * blockDim.x + threadIdx.x;
    if (id >= BATCH * 32 * NN) return;
    int b  = id / (32 * NN);
    int r  = id % (32 * NN);
    int oc = r / NN;
    int p  = r % NN;
    int py = p / WIDTH, px = p % WIDTH;
    float s = bias[oc];
    #pragma unroll
    for (int ic = 0; ic < 3; ++ic)
        #pragma unroll
        for (int ky = 0; ky < 3; ++ky) {
            int yy = py + ky - 1;
            if (yy < 0 || yy >= WIDTH) continue;
            #pragma unroll
            for (int kx = 0; kx < 3; ++kx) {
                int xx = px + kx - 1;
                if (xx < 0 || xx >= WIDTH) continue;
                s += w[((oc * 3 + ic) * 3 + ky) * 3 + kx] *
                     xf[(b * 3 + ic) * NN + yy * WIDTH + xx];
            }
        }
    s = s > 0.0f ? s : 0.0f;
    actOut[act_idx(b, py, px) + oc] = (bf16)s;
}

// ---------------------------------------------------------------------------
// Mid layers: 32->32 conv, implicit GEMM via V_WMMA_F32_16X16X32_BF16.
// One wave/block; M-tile = 16 pixels; K = 9 taps x 32 ic (1 WMMA per tap).
// A fragment = two aligned b128 loads from padded pixel-major activations
// (elements 0-7 -> ic 8*hi+0..7, 8-15 -> ic 16+8*hi+0..7 per ISA 7.12.2).
// B fragment = one aligned 32B load of pre-packed weights.
// ---------------------------------------------------------------------------
__global__ void conv_mid_wmma(const bf16* __restrict__ actIn,
                              bf16* __restrict__ actOut,
                              const bf16* __restrict__ wpk,    // [9][2][32][16]
                              const float* __restrict__ bias) {
    const int lane  = threadIdx.x;       // 0..31
    const int tileM = blockIdx.x;        // 0..80
    const int b     = blockIdx.y;
    const int mrow  = lane & 15;
    const int hi    = lane >> 4;

    v8f acc0 = {}; v8f acc1 = {};
    const int m  = tileM * 16 + mrow;
    const int py = m / WIDTH, px = m % WIDTH;

    #pragma unroll
    for (int t = 0; t < 9; ++t) {
        const int dy = t / 3 - 1, dx = t % 3 - 1;
        const int pix = act_idx(b, py + dy, px + dx);   // always in padded range
        union { v16bf v; v8bf h[2]; } A;
        A.h[0] = *(const v8bf*)(actIn + pix + 8 * hi);
        A.h[1] = *(const v8bf*)(actIn + pix + 16 + 8 * hi);

        v16bf b0 = *(const v16bf*)(wpk + ((t * 2 + 0) * 32 + lane) * 16);
        v16bf b1 = *(const v16bf*)(wpk + ((t * 2 + 1) * 32 + lane) * 16);
        acc0 = __builtin_amdgcn_wmma_f32_16x16x32_bf16(
                   false, A.v, false, b0, (short)0, acc0, false, false);
        acc1 = __builtin_amdgcn_wmma_f32_16x16x32_bf16(
                   false, A.v, false, b1, (short)0, acc1, false, false);
    }
    // D: element v -> M = v + 8*hi, N = lane&15. Bias + ReLU, store bf16.
    const float bi0 = bias[mrow], bi1 = bias[16 + mrow];
    #pragma unroll
    for (int v = 0; v < 8; ++v) {
        int mo  = tileM * 16 + v + 8 * hi;
        int pyo = mo / WIDTH, pxo = mo % WIDTH;
        int o   = act_idx(b, pyo, pxo);
        float f0 = acc0[v] + bi0;  f0 = f0 > 0 ? f0 : 0;
        float f1 = acc1[v] + bi1;  f1 = f1 > 0 ? f1 : 0;
        actOut[o + mrow]      = (bf16)f0;
        actOut[o + 16 + mrow] = (bf16)f1;
    }
}

// ---------------------------------------------------------------------------
// Output layer: 32->6 conv via WMMA (N padded to 16), f32 features out.
// ---------------------------------------------------------------------------
__global__ void conv_out_wmma(const bf16* __restrict__ actIn,
                              float* __restrict__ feat,        // [B,6,NN]
                              const bf16* __restrict__ wpk,    // [9][32][16]
                              const float* __restrict__ bias) {
    const int lane  = threadIdx.x;
    const int tileM = blockIdx.x;
    const int b     = blockIdx.y;
    const int mrow  = lane & 15;
    const int hi    = lane >> 4;

    v8f acc = {};
    const int m  = tileM * 16 + mrow;
    const int py = m / WIDTH, px = m % WIDTH;

    #pragma unroll
    for (int t = 0; t < 9; ++t) {
        const int dy = t / 3 - 1, dx = t % 3 - 1;
        const int pix = act_idx(b, py + dy, px + dx);
        union { v16bf v; v8bf h[2]; } A;
        A.h[0] = *(const v8bf*)(actIn + pix + 8 * hi);
        A.h[1] = *(const v8bf*)(actIn + pix + 16 + 8 * hi);
        v16bf bb = *(const v16bf*)(wpk + (t * 32 + lane) * 16);
        acc = __builtin_amdgcn_wmma_f32_16x16x32_bf16(
                  false, A.v, false, bb, (short)0, acc, false, false);
    }
    if (mrow < 6) {
        float bi = bias[mrow];
        #pragma unroll
        for (int v = 0; v < 8; ++v) {
            int mo = tileM * 16 + v + 8 * hi;
            feat[(b * 6 + mrow) * NN + mo] = acc[v] + bi;
        }
    }
}

// ---------------------------------------------------------------------------
// Edge weights: w_e = exp(-sum_f (f[i0]-f[i1])^2 / sigma^2)
// ---------------------------------------------------------------------------
__global__ void edge_weight_kernel(const float* __restrict__ feat,
                                   const int* __restrict__ idx0,
                                   const int* __restrict__ idx1,
                                   float* __restrict__ wE) {
    int id = blockIdx.x * blockDim.x + threadIdx.x;
    if (id >= BATCH * NE) return;
    int b = id / NE, e = id % NE;
    int i0 = idx0[e], i1 = idx1[e];
    float s = 0.0f;
    #pragma unroll
    for (int f = 0; f < 6; ++f) {
        float d = feat[(b * 6 + f) * NN + i0] - feat[(b * 6 + f) * NN + i1];
        s += d * d;
    }
    wE[b * NE + e] = expf(-s * SIGMA2_INV);
}

// ---------------------------------------------------------------------------
// GTV solver: per (b,c) workgroup; 9 outer reweight rounds, each solving
// (I + u*L) x = y by warm-started CG, entirely LDS-resident (~37 KB/WG of
// the 320 KB/WGP LDS). 5-point stencil SpMV, tree-reduced dot products.
// ---------------------------------------------------------------------------
__device__ __forceinline__ float block_reduce(float v, float* red) {
    int tid = threadIdx.x;
    red[tid] = v;
    __syncthreads();
    for (int off = 128; off > 0; off >>= 1) {
        if (tid < off) red[tid] += red[tid + off];
        __syncthreads();
    }
    float r = red[0];
    __syncthreads();
    return r;
}

__device__ __forceinline__ void apply_A(const float* __restrict__ xv,
                                        float* __restrict__ out,
                                        const float* __restrict__ lwh,
                                        const float* __restrict__ lwv,
                                        float u) {
    for (int n = threadIdx.x; n < NN; n += 256) {
        int py = n / WIDTH, px = n % WIDTH;
        float xc = xv[n];
        float acc = 0.0f;
        if (px > 0)         acc += lwh[py * (WIDTH - 1) + px - 1] * (xc - xv[n - 1]);
        if (px < WIDTH - 1) acc += lwh[py * (WIDTH - 1) + px]     * (xc - xv[n + 1]);
        if (py > 0)         acc += lwv[n - WIDTH]                 * (xc - xv[n - WIDTH]);
        if (py < WIDTH - 1) acc += lwv[n]                         * (xc - xv[n + WIDTH]);
        out[n] = xc + u * acc;
    }
}

__global__ void __launch_bounds__(256)
gtv_solver(const float* __restrict__ xf,
           const float* __restrict__ u_param,
           const float* __restrict__ wE,
           float* __restrict__ out) {
    __shared__ float yv[NN], sv[NN], rv[NN], pv[NN], apv[NN];
    __shared__ float lwh[EH], lwv[EV], red[256];

    const int tid = threadIdx.x;
    const int bc  = blockIdx.x;          // 0..5
    const int b   = bc / CH, c = bc % CH;
    const float u = fminf(fmaxf(u_param[0], 0.001f), 1.0f);
    const float* wEb = wE + b * NE;

    for (int n = tid; n < NN; n += 256) yv[n] = xf[(b * CH + c) * NN + n];
    __syncthreads();

    for (int it = 0; it < NITERS; ++it) {
        for (int e = tid; e < EH; e += 256) {
            int i = e / (WIDTH - 1), j = e % (WIDTH - 1);
            int n0 = i * WIDTH + j;
            float z = fabsf(yv[n0] - yv[n0 + 1]);
            lwh[e] = wEb[e] / fmaxf(z, ZMAXV);
        }
        for (int e = tid; e < EV; e += 256) {     // n0 == e for vertical edges
            float z = fabsf(yv[e] - yv[e + WIDTH]);
            lwv[e] = wEb[EH + e] / fmaxf(z, ZMAXV);
        }
        __syncthreads();

        for (int n = tid; n < NN; n += 256) sv[n] = yv[n];
        __syncthreads();
        apply_A(sv, apv, lwh, lwv, u);
        __syncthreads();
        float part = 0.0f;
        for (int n = tid; n < NN; n += 256) {
            float rr = yv[n] - apv[n];
            rv[n] = rr; pv[n] = rr;
            part += rr * rr;
        }
        float rs  = block_reduce(part, red);
        float tol = rs * 1e-12f + 1e-30f;

        for (int k = 0; k < CG_MAX; ++k) {
            if (rs <= tol) break;                 // uniform: rs from shared reduce
            apply_A(pv, apv, lwh, lwv, u);
            __syncthreads();
            float pa = 0.0f;
            for (int n = tid; n < NN; n += 256) pa += pv[n] * apv[n];
            float pAp = block_reduce(pa, red);
            float alpha = rs / pAp;
            float rn = 0.0f;
            for (int n = tid; n < NN; n += 256) {
                sv[n] += alpha * pv[n];
                float rr = rv[n] - alpha * apv[n];
                rv[n] = rr;
                rn += rr * rr;
            }
            __syncthreads();
            float rsnew = block_reduce(rn, red);
            float beta = rsnew / rs;
            for (int n = tid; n < NN; n += 256) pv[n] = rv[n] + beta * pv[n];
            rs = rsnew;
            __syncthreads();
        }
        for (int n = tid; n < NN; n += 256) yv[n] = sv[n];
        __syncthreads();
    }
    for (int n = tid; n < NN; n += 256) out[(b * CH + c) * NN + n] = yv[n];
}

// ---------------------------------------------------------------------------
// Workspace layout (bytes, 64B-aligned blocks):
//   actA   : 0      B*PNN*32 bf16 = 184832
//   actB   : 184832                 184832
//   wMidPk : 369664 5*9216 bf16   =  92160
//   wOutPk : 461824 9*512 bf16    =   9216
//   feat   : 471040 B*6*NN f32    =  62208
//   wE     : 533248 B*NE f32      =  20160   total 553408 B
// ---------------------------------------------------------------------------
extern "C" void kernel_launch(void* const* d_in, const int* in_sizes, int n_in,
                              void* d_out, int out_size, void* d_ws, size_t ws_size,
                              hipStream_t stream) {
    const float* xf     = (const float*)d_in[0];
    const float* up     = (const float*)d_in[1];
    // d_in[2] = dense incidence H, unused (stencil form used instead)
    const float* cw_in  = (const float*)d_in[3];
    const float* b_in   = (const float*)d_in[4];
    const float* cw_mid = (const float*)d_in[5];
    const float* b_mid  = (const float*)d_in[6];
    const float* cw_out = (const float*)d_in[7];
    const float* b_out  = (const float*)d_in[8];
    const int*   idx0   = (const int*)d_in[9];
    const int*   idx1   = (const int*)d_in[10];
    float* outp = (float*)d_out;

    char* ws = (char*)d_ws;
    bf16*  actA   = (bf16*)(ws + 0);
    bf16*  actB   = (bf16*)(ws + 184832);
    bf16*  wMidPk = (bf16*)(ws + 369664);
    bf16*  wOutPk = (bf16*)(ws + 461824);
    float* feat   = (float*)(ws + 471040);
    float* wE     = (float*)(ws + 533248);

    // 1) clear both activation buffers (zero halos); 2 buffers contiguous
    const int zwords = (2 * 184832) / 4;
    zero_kernel<<<(zwords + 255) / 256, 256, 0, stream>>>((unsigned int*)ws, zwords);

    // 2) pre-pack weights into WMMA B-fragment order (bf16)
    pack_mid_w<<<(5 * 9216 + 255) / 256, 256, 0, stream>>>(cw_mid, wMidPk);
    pack_out_w<<<(9 * 512 + 255) / 256, 256, 0, stream>>>(cw_out, wOutPk);

    // 3) CNN
    conv_in_kernel<<<(BATCH * 32 * NN + 255) / 256, 256, 0, stream>>>(
        xf, cw_in, b_in, actA);

    bf16* pin = actA; bf16* pout = actB;
    for (int l = 0; l < 5; ++l) {
        conv_mid_wmma<<<dim3(NN / 16, BATCH), 32, 0, stream>>>(
            pin, pout, wMidPk + l * 9216, b_mid + l * 32);
        bf16* t = pin; pin = pout; pout = t;
    }
    conv_out_wmma<<<dim3(NN / 16, BATCH), 32, 0, stream>>>(
        pin, feat, wOutPk, b_out);

    // 4) edge weights + 5) GTV solve
    edge_weight_kernel<<<(BATCH * NE + 255) / 256, 256, 0, stream>>>(
        feat, idx0, idx1, wE);
    gtv_solver<<<BATCH * CH, 256, 0, stream>>>(xf, up, wE, outp);
}
